// LocalSpatialEncoding_8950711846139
// MI455X (gfx1250) — matrix operand order repaired and checked by reading, not compiled, not run on hardware
//
#include <hip/hip_runtime.h>
#include <math.h>

// CDNA5 (gfx1250) wave32 WMMA types
typedef __attribute__((ext_vector_type(16))) _Float16 v16h;
typedef __attribute__((ext_vector_type(8)))  float    v8f;
typedef __attribute__((ext_vector_type(2)))  float    f2v;

#define WAVES_PER_BLOCK 8
#define LDS_ROW 34            // row stride in halfs (17-bank step -> conflict-free gather)
#define TILE_HALFS (16 * LDS_ROW)

union Frag16 { v16h h; unsigned u[8]; _Float16 e[16]; };

// Gather a 16x32 f16 A-fragment from a row-major LDS tile, per ISA 7.12.2:
// lanes 0-15: VGPR0-3 = K0..7, VGPR4-7 = K16..23 (row M = lane)
// lanes 16-31: VGPR0-3 = K8..15, VGPR4-7 = K24..31 (row M = lane-16)
__device__ __forceinline__ v16h load_a_frag_lds(const _Float16* tile, int lane) {
    const int m = lane & 15, half = lane >> 4;
    Frag16 a;
    const _Float16* row = tile + m * LDS_ROW;
#pragma unroll
    for (int v = 0; v < 8; ++v) {
        const int k = (v < 4) ? (2 * v + 8 * half) : (16 + 2 * (v - 4) + 8 * half);
        a.u[v] = *(const unsigned*)(row + k);   // 2 halfs per VGPR, 4B aligned
    }
    return a.h;
}

// Build a 32x16 f16 B-fragment (cols [colBase, colBase+16)) from row-major f32 W.
// Per ISA B layout (col = lane&15): VGPRv holds K = (v<4 ? 2v : 16+2(v-4)) + 8*(lane>>4) + {0,1}.
// Rows >= `rows` are zero-padded (matches zero-padded K in A).
__device__ __forceinline__ v16h load_b_frag_global(const float* __restrict__ W, int ld,
                                                   int rows, int colBase, int lane) {
    const int c = colBase + (lane & 15);
    const int half = lane >> 4;
    Frag16 b;
#pragma unroll
    for (int v = 0; v < 8; ++v) {
        const int k0 = (v < 4) ? (2 * v + 8 * half) : (16 + 2 * (v - 4) + 8 * half);
        const float f0 = (k0     < rows) ? W[k0 * ld + c]       : 0.0f;
        const float f1 = (k0 + 1 < rows) ? W[(k0 + 1) * ld + c] : 0.0f;
        b.e[2 * v]     = (_Float16)f0;
        b.e[2 * v + 1] = (_Float16)f1;
    }
    return b.h;
}

// Splat one bias value into a C-fragment: C/D layout keeps one column (=one
// bias element) per lane across all 8 row-VGPRs, so bias rides in the WMMA add.
__device__ __forceinline__ v8f splat_c(float x) {
    v8f c;
#pragma unroll
    for (int v = 0; v < 8; ++v) c[v] = x;
    return c;
}

__global__ void lse_wmma_kernel(const float* __restrict__ coords,
                                const float* __restrict__ features,
                                const float* __restrict__ W1,
                                const float* __restrict__ b1,
                                const float* __restrict__ W2,
                                const float* __restrict__ b2,
                                const int*   __restrict__ nbr,
                                float* __restrict__ out,
                                int N)
{
    __shared__ _Float16 smem[WAVES_PER_BLOCK][2][TILE_HALFS];

    const int lane = threadIdx.x & 31;
    const int wave = threadIdx.x >> 5;
    const int m    = lane & 15;     // row within tile / col within C-D layout
    const int half = lane >> 4;

    _Float16* A1 = smem[wave][0];   // 16x32 concat tile (f16, zero padded K10..31)
    _Float16* HT = smem[wave][1];   // 16x32 hidden tile (f16)

    // One-time zero init of staging tiles (padding stays zero forever).
    for (int i = lane; i < TILE_HALFS; i += 32) {
        A1[i] = (_Float16)0.0f;
        HT[i] = (_Float16)0.0f;
    }

    // Constant weight fragments, amortized over the grid-stride loop.
    const v16h B1a = load_b_frag_global(W1, 32, 10, 0,  lane);  // W1 (10,32) -> K padded to 32
    const v16h B1b = load_b_frag_global(W1, 32, 10, 16, lane);
    v16h B2t[4];
#pragma unroll
    for (int t = 0; t < 4; ++t) B2t[t] = load_b_frag_global(W2, 64, 32, 16 * t, lane);

    // Bias pre-loaded into C accumulators (D = A*B + bias from the matrix unit).
    const v8f C1a = splat_c(b1[m]);
    const v8f C1b = splat_c(b1[16 + m]);
    v8f C2[4];
#pragma unroll
    for (int t = 0; t < 4; ++t) C2[t] = splat_c(b2[16 * t + m]);

    const int wavesTotal = gridDim.x * WAVES_PER_BLOCK;

    for (int n = blockIdx.x * WAVES_PER_BLOCK + wave; n < N; n += wavesTotal) {
        // ---- build 16 concat rows (K=16 neighbors == exactly one WMMA M-tile) ----
        const float ox = coords[3 * n + 0];
        const float oy = coords[3 * n + 1];
        const float oz = coords[3 * n + 2];
        if (half == 0) {
            const int idx = nbr[(size_t)n * 16 + m];
            const float nx = coords[3 * idx + 0];
            const float ny = coords[3 * idx + 1];
            const float nz = coords[3 * idx + 2];
            const float rx = ox - nx, ry = oy - ny, rz = oz - nz;
            const float dist = sqrtf(rx * rx + ry * ry + rz * rz);
            _Float16* row = A1 + m * LDS_ROW;
            row[0] = (_Float16)ox; row[1] = (_Float16)oy; row[2] = (_Float16)oz;
            row[3] = (_Float16)nx; row[4] = (_Float16)ny; row[5] = (_Float16)nz;
            row[6] = (_Float16)rx; row[7] = (_Float16)ry; row[8] = (_Float16)rz;
            row[9] = (_Float16)dist;            // cols 10..31 stay zero from init
        }
        // DS ops are in-order within a wave; wait out the stores explicitly anyway.
        asm volatile("s_wait_dscnt 0" ::: "memory");

        // ---- matmul 1: (16x32) x (32x32) + b1 -> h (16x32), ReLU ----
        const v16h a1 = load_a_frag_lds(A1, lane);
        v8f h0 = __builtin_amdgcn_wmma_f32_16x16x32_f16(false, a1, false, B1a,
                                                        (short)0, C1a, false, false);
        v8f h1 = __builtin_amdgcn_wmma_f32_16x16x32_f16(false, a1, false, B1b,
                                                        (short)0, C1b, false, false);
#pragma unroll
        for (int v = 0; v < 8; ++v) {          // C/D layout: VGPRv -> row v + 8*half, col m
            const int row = v + 8 * half;
            const float x0 = h0[v] > 0.0f ? h0[v] : 0.0f;
            const float x1 = h1[v] > 0.0f ? h1[v] : 0.0f;
            HT[row * LDS_ROW + m]      = (_Float16)x0;
            HT[row * LDS_ROW + 16 + m] = (_Float16)x1;
        }
        asm volatile("s_wait_dscnt 0" ::: "memory");

        // ---- matmul 2: (16x32) x (32x64) + b2 -> enc (16x64) ----
        const v16h a2 = load_a_frag_lds(HT, lane);
        v8f enc[4];
#pragma unroll
        for (int t = 0; t < 4; ++t) {
            enc[t] = __builtin_amdgcn_wmma_f32_16x16x32_f16(false, a2, false, B2t[t],
                                                            (short)0, C2[t], false, false);
        }

        // ---- output: out[n, k, 0:64] = enc ; out[n, k, 64:128] = features[n] ----
        // Streamed once, never re-read: non-temporal stores keep the big L2 for
        // the reused coords/features/weights instead of 819MB of output.
        float* outp = out + (size_t)n * (16 * 128);
        const f2v f2 = __builtin_nontemporal_load(
            (const f2v*)(features + (size_t)n * 64) + lane) ;

#pragma unroll
        for (int t = 0; t < 4; ++t) {
#pragma unroll
            for (int v = 0; v < 8; ++v) {
                const int row = v + 8 * half;
                __builtin_nontemporal_store(enc[t][v], outp + row * 128 + 16 * t + m);
            }
        }
#pragma unroll
        for (int r = 0; r < 16; ++r) {
            __builtin_nontemporal_store(f2, (f2v*)(outp + r * 128 + 64) + lane);
        }
    }
}

extern "C" void kernel_launch(void* const* d_in, const int* in_sizes, int n_in,
                              void* d_out, int out_size, void* d_ws, size_t ws_size,
                              hipStream_t stream) {
    const float* coords   = (const float*)d_in[0];
    const float* features = (const float*)d_in[1];
    const float* W1       = (const float*)d_in[2];
    const float* b1       = (const float*)d_in[3];
    const float* W2       = (const float*)d_in[4];
    const float* b2       = (const float*)d_in[5];
    const int*   nbr      = (const int*)d_in[6];
    float*       out      = (float*)d_out;

    const int N = in_sizes[0] / 3;   // coords is (N,3)

    int blocks = (N + WAVES_PER_BLOCK - 1) / WAVES_PER_BLOCK;
    if (blocks > 1024) blocks = 1024;        // grid-stride loop amortizes weight-fragment setup
    if (blocks < 1) blocks = 1;

    lse_wmma_kernel<<<blocks, WAVES_PER_BLOCK * 32, 0, stream>>>(
        coords, features, W1, b1, W2, b2, nbr, out, N);
}
